// AdvancedTransformerEncoderBlock_89773406421449
// MI455X (gfx1250) — matrix operand order repaired and checked by reading, not compiled, not run on hardware
//
#include <hip/hip_runtime.h>

// ---------------------------------------------------------------------------
// Types / feature probes
// ---------------------------------------------------------------------------
typedef _Float16 h16;
typedef __attribute__((ext_vector_type(16))) _Float16 v16h;
typedef __attribute__((ext_vector_type(8)))  _Float16 v8h;
typedef __attribute__((ext_vector_type(4)))  _Float16 v4h;
typedef __attribute__((ext_vector_type(8)))  float    v8f;
typedef int v4i __attribute__((__vector_size__(4 * sizeof(int))));

#define AS1 __attribute__((address_space(1)))
#define AS3 __attribute__((address_space(3)))

#if defined(__AMDGCN__) && __has_builtin(__builtin_amdgcn_global_load_async_to_lds_b128) && \
    __has_builtin(__builtin_amdgcn_s_wait_asynccnt)
#define HAVE_ASYNC 1
#else
#define HAVE_ASYNC 0
#endif

__device__ __forceinline__ v16h mk16(v8h lo, v8h hi) {
  v16h r;
#pragma unroll
  for (int i = 0; i < 8; ++i) { r[i] = lo[i]; r[8 + i] = hi[i]; }
  return r;
}

__device__ __forceinline__ v8f wmma16(v16h a, v16h b, v8f c) {
  // D = A(16x32 f16) * B(32x16 f16) + C(16x16 f32)
  return __builtin_amdgcn_wmma_f32_16x16x32_f16(false, a, false, b, (short)0, c,
                                                false, false);
}

// ---------------------------------------------------------------------------
// fp32 -> fp16 conversion (weights)
// ---------------------------------------------------------------------------
__global__ __launch_bounds__(256)
void cvt_f32_f16(const float* __restrict__ in, h16* __restrict__ out, int n) {
  int i = (blockIdx.x * 256 + threadIdx.x) * 4;
  if (i < n) {
    float4 v = *(const float4*)(in + i);
    v4h o;
    o[0] = (h16)v.x; o[1] = (h16)v.y; o[2] = (h16)v.z; o[3] = (h16)v.w;
    *(v4h*)(out + i) = o;
  }
}

// ---------------------------------------------------------------------------
// LayerNorm: one block per row (D = 1024), writes f16 activations
// ---------------------------------------------------------------------------
__global__ __launch_bounds__(256)
void layernorm_f16(const float* __restrict__ x, const float* __restrict__ g,
                   const float* __restrict__ bb, h16* __restrict__ y) {
  const int D = 1024;
  __shared__ float red[256];
  int row = blockIdx.x, tid = threadIdx.x;
  const float* xr = x + (size_t)row * D;
  float v[4]; float s = 0.f;
#pragma unroll
  for (int i = 0; i < 4; ++i) { v[i] = xr[tid + 256 * i]; s += v[i]; }
  red[tid] = s; __syncthreads();
  for (int off = 128; off > 0; off >>= 1) {
    if (tid < off) red[tid] += red[tid + off];
    __syncthreads();
  }
  float mean = red[0] * (1.0f / D);
  __syncthreads();
  float q = 0.f;
#pragma unroll
  for (int i = 0; i < 4; ++i) { float d = v[i] - mean; q += d * d; }
  red[tid] = q; __syncthreads();
  for (int off = 128; off > 0; off >>= 1) {
    if (tid < off) red[tid] += red[tid + off];
    __syncthreads();
  }
  float rs = rsqrtf(red[0] * (1.0f / D) + 1e-5f);
  h16* yr = y + (size_t)row * D;
#pragma unroll
  for (int i = 0; i < 4; ++i) {
    int c = tid + 256 * i;
    yr[c] = (h16)((v[i] - mean) * rs * g[c] + bb[c]);
  }
}

// ---------------------------------------------------------------------------
// Tiled WMMA GEMM: C[R,N] = A[R,K] * W[K,N] (+ fused epilogues)
// Block 256 thr (8 waves, 4x2 wave grid). Tile 128x128, K-step 32.
// A tile staged via GLOBAL_LOAD_ASYNC_TO_LDS_B128 (ASYNCcnt) when available.
// LDS rows padded to 40 f16 (80B) -> conflict-free 16-lane fragment gathers.
// ---------------------------------------------------------------------------
enum { EPI_PLAIN = 0, EPI_BIAS = 1, EPI_RESID = 2, EPI_SWIGLU = 3, EPI_DOWN = 4 };

template <int EPI>
__global__ __launch_bounds__(256)
void gemm_f16(const h16* __restrict__ A, const h16* __restrict__ Wm,
              int N, int K,
              const float* __restrict__ bias, const float* __restrict__ aux,
              const float* __restrict__ resid, float* __restrict__ outf,
              h16* __restrict__ outh) {
  constexpr int ASTR = 40;  // padded LDS row stride (f16 elems)
  __shared__ __align__(16) h16 sA[128 * ASTR];  // [m][k]
  __shared__ __align__(16) h16 sB[128 * ASTR];  // [n][k] (transposed on store)
  const int tid = threadIdx.x, lane = tid & 31, w = tid >> 5;
  const int hl = lane >> 4, lr = lane & 15;
  const int wm = w & 3, wn = w >> 2;
  const int m0 = blockIdx.y * 128, n0 = blockIdx.x * 128;

  v8f acc[2][4];
#pragma unroll
  for (int mi = 0; mi < 2; ++mi)
#pragma unroll
    for (int nt = 0; nt < 4; ++nt)
#pragma unroll
      for (int e = 0; e < 8; ++e) acc[mi][nt][e] = 0.f;

  for (int k0 = 0; k0 < K; k0 += 32) {
    __syncthreads();
#if HAVE_ASYNC
    {  // A tile 128x32: async DMA global->LDS, 2 x 16B chunks per thread
#pragma unroll
      for (int cc = 0; cc < 2; ++cc) {
        int c = tid * 2 + cc;          // 0..511 chunks of 8 f16
        int row = c >> 2, ko = (c & 3) * 8;
        __builtin_amdgcn_global_load_async_to_lds_b128(
            (AS1 v4i*)(A + (size_t)(m0 + row) * K + k0 + ko),
            (AS3 v4i*)(sA + row * ASTR + ko), 0, 0);
      }
    }
#else
    {  // A tile 128x32: each thread two 16B row-chunks via VGPR staging
      int r = tid >> 2, kc = (tid & 3) * 8;
#pragma unroll
      for (int rr = 0; rr < 2; ++rr) {
        int row = r + rr * 64;
        v8h d = *(const v8h*)(A + (size_t)(m0 + row) * K + k0 + kc);
        *(v8h*)(sA + row * ASTR + kc) = d;
      }
    }
#endif
    {  // W tile 32x128, transposed into sB[n][k]
#pragma unroll
      for (int it = 0; it < 2; ++it) {
        int task = tid + it * 256;
        int k = task & 31;
        int n = (task >> 5) * 8;
        v8h d = *(const v8h*)(Wm + (size_t)(k0 + k) * N + n0 + n);
#pragma unroll
        for (int i = 0; i < 8; ++i) sB[(n + i) * ASTR + k] = d[i];
      }
    }
    if (k0 + 32 < K) {  // codegen-confirmed global_prefetch_b8 path
      __builtin_prefetch(A + (size_t)(m0 + (tid >> 2)) * K + k0 + 32, 0, 0);
      __builtin_prefetch(Wm + (size_t)(k0 + 32 + (tid & 31)) * N + n0, 0, 0);
    }
#if HAVE_ASYNC
    __builtin_amdgcn_s_wait_asynccnt(0);  // A-tile DMA complete before barrier
#endif
    __syncthreads();

    v16h af[2], bfr[4];
#pragma unroll
    for (int mi = 0; mi < 2; ++mi) {
      // A frag: row = lane&15; K = 16*hl + e (contiguous)
      const h16* p = sA + (wm * 32 + mi * 16 + lr) * ASTR + 16 * hl;
      af[mi] = mk16(*(const v8h*)p, *(const v8h*)(p + 8));
    }
#pragma unroll
    for (int nt = 0; nt < 4; ++nt) {
      // B frag: col = lane&15; K = 16*hl + e (contiguous)
      const h16* p = sB + (wn * 64 + nt * 16 + lr) * ASTR + 16 * hl;
      bfr[nt] = mk16(*(const v8h*)p, *(const v8h*)(p + 8));
    }
#pragma unroll
    for (int mi = 0; mi < 2; ++mi)
#pragma unroll
      for (int nt = 0; nt < 4; ++nt)
        acc[mi][nt] = wmma16(af[mi], bfr[nt], acc[mi][nt]);
  }

  // Epilogue: C layout M = v + 8*hl, N = lane&15
#pragma unroll
  for (int mi = 0; mi < 2; ++mi)
#pragma unroll
    for (int nt = 0; nt < 4; ++nt)
#pragma unroll
      for (int v = 0; v < 8; ++v) {
        int row = m0 + wm * 32 + mi * 16 + v + 8 * hl;
        int col = n0 + wn * 64 + nt * 16 + lr;
        size_t idx = (size_t)row * N + col;
        float val = acc[mi][nt][v];
        if (EPI == EPI_PLAIN) outf[idx] = val;
        if (EPI == EPI_BIAS) outf[idx] = val + bias[col];
        if (EPI == EPI_RESID) outf[idx] = val + resid[idx];
        if (EPI == EPI_SWIGLU) {
          float u = val + bias[col];
          float sg = 1.f / (1.f + __expf(-u));
          outh[idx] = (h16)(aux[idx] * u * sg);
        }
        if (EPI == EPI_DOWN) outf[idx] = val + bias[col] + resid[idx];
      }
}

// ---------------------------------------------------------------------------
// RoPE + head split + f16 repack. qkv f32 [B*S, 3*1024] ->
//   q,k f16 [B,H,S,64] (q pre-scaled by 1/sqrt(hd)), v transposed [B,H,64,S]
// ---------------------------------------------------------------------------
__global__ __launch_bounds__(256)
void rope_split(const float* __restrict__ qkv, h16* __restrict__ q,
                h16* __restrict__ k, h16* __restrict__ vt) {
  int gid = blockIdx.x * 256 + threadIdx.x;  // B*H*S*32 total
  int d2 = gid & 31;
  int s  = (gid >> 5) & 2047;
  int hh = (gid >> 16) & 15;
  int b  = gid >> 20;
  const float* base = qkv + (size_t)(b * 2048 + s) * 3072 + hh * 64;
  float q0 = base[d2],        q1 = base[d2 + 32];
  float k0 = base[1024 + d2], k1 = base[1024 + d2 + 32];
  float v0 = base[2048 + d2], v1 = base[2048 + d2 + 32];
  // inv_freq = 10000^{-d2/32} = 2^{-d2*log2(10000)/32}
  float fr = exp2f((float)d2 * (-13.287712379549449f / 32.0f));
  float ang = (float)s * fr;
  float sn, cs;
  __sincosf(ang, &sn, &cs);
  size_t o = ((size_t)(b * 16 + hh) * 2048 + s) * 64;
  q[o + d2]      = (h16)((q0 * cs - q1 * sn) * 0.125f);
  q[o + d2 + 32] = (h16)((q1 * cs + q0 * sn) * 0.125f);
  k[o + d2]      = (h16)(k0 * cs - k1 * sn);
  k[o + d2 + 32] = (h16)(k1 * cs + k0 * sn);
  size_t vb = (size_t)(b * 16 + hh) * 64;
  vt[(vb + d2) * 2048 + s]      = (h16)v0;
  vt[(vb + d2 + 32) * 2048 + s] = (h16)v1;
}

// ---------------------------------------------------------------------------
// Flash-style local-causal attention, band width 257.
// Block = 4 waves; wave w owns 16 query rows i0 = 64*blockIdx.y + 16*w.
// Q*K^T and P*V via WMMA straight from global (layouts match fragments);
// P gets a C->A layout swap through a 1KB per-wave LDS bounce.
// ---------------------------------------------------------------------------
__global__ __launch_bounds__(128)
void attn_local(const h16* __restrict__ Q, const h16* __restrict__ K,
                const h16* __restrict__ Vt, h16* __restrict__ O) {
  constexpr int S = 2048, HD = 64, H = 16, WIN = 256;
  __shared__ __align__(16) h16 pbuf[4][16 * 32];
  int tid = threadIdx.x, w = tid >> 5, lane = tid & 31;
  int hl = lane >> 4, lr = lane & 15;
  int bh = blockIdx.x, b = bh >> 4, hh = bh & 15;
  int i0 = blockIdx.y * 64 + w * 16;
  const h16* Qb = Q + (size_t)bh * S * HD;
  const h16* Kb = K + (size_t)bh * S * HD;
  const h16* Vb = Vt + (size_t)bh * HD * S;
  h16* pb = pbuf[w];

  // Q fragments (A-layout): row = lane&15; two hd-halves
  v16h qf[2];
  {
    const h16* qr = Qb + (size_t)(i0 + lr) * HD;
#pragma unroll
    for (int fi = 0; fi < 2; ++fi) {
      const h16* p = qr + fi * 32 + 16 * hl;
      qf[fi] = mk16(*(const v8h*)p, *(const v8h*)(p + 8));
    }
  }

  float mrow[8], srow[8];
  v8f oacc[4];
#pragma unroll
  for (int v = 0; v < 8; ++v) { mrow[v] = -1e30f; srow[v] = 0.f; }
#pragma unroll
  for (int nt = 0; nt < 4; ++nt)
#pragma unroll
    for (int e = 0; e < 8; ++e) oacc[nt][e] = 0.f;

  int jlo = i0 - WIN; if (jlo < 0) jlo = 0; jlo &= ~31;
  for (int j0 = jlo; j0 <= i0 + 15; j0 += 32) {
    v8f c[2];
#pragma unroll
    for (int t = 0; t < 2; ++t) {
      int jt = j0 + 16 * t;
      int j = jt + lr; if (j > S - 1) j = S - 1;  // clamp; masked cols -> p=0
      // K^T B-frag: col = key j; K-dim = feature, contiguous in memory
      const h16* kr = Kb + (size_t)j * HD + 16 * hl;
      v16h kf0 = mk16(*(const v8h*)kr,        *(const v8h*)(kr + 8));
      v16h kf1 = mk16(*(const v8h*)(kr + 32), *(const v8h*)(kr + 40));
      v8f z;
#pragma unroll
      for (int e = 0; e < 8; ++e) z[e] = 0.f;
      z = wmma16(qf[0], kf0, z);
      z = wmma16(qf[1], kf1, z);
#pragma unroll
      for (int v = 0; v < 8; ++v) {   // mask: causal + local band
        int i = i0 + v + 8 * hl;
        int jj = jt + lr;
        bool ok = (jj <= i) && (i - jj <= WIN);
        z[v] = ok ? z[v] : -1e30f;
      }
      c[t] = z;
    }
    // streaming softmax update (row lives across one 16-lane half)
#pragma unroll
    for (int v = 0; v < 8; ++v) {
      float mx = fmaxf(c[0][v], c[1][v]);
#pragma unroll
      for (int off = 8; off > 0; off >>= 1)
        mx = fmaxf(mx, __shfl_xor(mx, off, 32));
      float mnew = fmaxf(mrow[v], mx);
      float resc = __expf(mrow[v] - mnew);
      mrow[v] = mnew;
      float p0 = __expf(c[0][v] - mnew);
      float p1 = __expf(c[1][v] - mnew);
      float ps = p0 + p1;
#pragma unroll
      for (int off = 8; off > 0; off >>= 1) ps += __shfl_xor(ps, off, 32);
      srow[v] = srow[v] * resc + ps;
#pragma unroll
      for (int nt = 0; nt < 4; ++nt) oacc[nt][v] *= resc;
      // stash P in C-layout positions (row = v+8*hl, col = 16t+lr)
      pb[(v + 8 * hl) * 32 + lr]      = (h16)p0;
      pb[(v + 8 * hl) * 32 + 16 + lr] = (h16)p1;
    }
    // reload P as A-fragment (same-wave DS ops are in-order)
    v16h pf;
    {
      const h16* p = pb + lr * 32 + 16 * hl;
      pf = mk16(*(const v8h*)p, *(const v8h*)(p + 8));
    }
    // P*V: Vt B-frags load contiguous along S
#pragma unroll
    for (int nt = 0; nt < 4; ++nt) {
      int f = 16 * nt + lr;
      int cs = j0 + 16 * hl; if (cs > S - 16) cs = S - 16;
      const h16* vr = Vb + (size_t)f * S + cs;
      v16h vf = mk16(*(const v8h*)vr, *(const v8h*)(vr + 8));
      oacc[nt] = wmma16(pf, vf, oacc[nt]);
    }
  }
  // finalize: O[b, i, h, f] so that flat [B*S, 1024] feeds the out-proj GEMM
#pragma unroll
  for (int v = 0; v < 8; ++v) {
    float inv = 1.f / srow[v];
#pragma unroll
    for (int nt = 0; nt < 4; ++nt) {
      int i = i0 + v + 8 * hl;
      int f = 16 * nt + lr;
      O[((size_t)(b * S + i) * H + hh) * HD + f] = (h16)(oacc[nt][v] * inv);
    }
  }
}

// ---------------------------------------------------------------------------
// Host-side orchestration
// ---------------------------------------------------------------------------
extern "C" void kernel_launch(void* const* d_in, const int* in_sizes, int n_in,
                              void* d_out, int out_size, void* d_ws,
                              size_t ws_size, hipStream_t stream) {
  const int S = 2048, D = 1024, F = 4096, R = 2 * S;  // B=2, H=16, hd=64
  const float* x      = (const float*)d_in[0];
  const float* w_qkv  = (const float*)d_in[1];
  const float* w_out  = (const float*)d_in[2];
  const float* g1     = (const float*)d_in[3];
  const float* b1     = (const float*)d_in[4];
  const float* g2     = (const float*)d_in[5];
  const float* b2     = (const float*)d_in[6];
  const float* w_gate = (const float*)d_in[7];
  const float* b_gate = (const float*)d_in[8];
  const float* w_up   = (const float*)d_in[9];
  const float* b_up   = (const float*)d_in[10];
  const float* w_down = (const float*)d_in[11];
  const float* b_down = (const float*)d_in[12];
  float* out = (float*)d_out;

  // Workspace layout (peak ~152 MB, dead regions reused across phases)
  char* ws = (char*)d_ws;
  const size_t MB = (size_t)1 << 20;
  float* x1      = (float*)(ws + 0);          // 16 MB, persists
  h16*   ybf     = (h16*)(ws + 16 * MB);      // 8 MB (ln1 then ln2)
  h16*   wqkvh   = (h16*)(ws + 24 * MB);      // 6 MB
  h16*   wouth   = (h16*)(ws + 30 * MB);      // 2 MB
  h16*   wgateh  = (h16*)(ws + 32 * MB);      // 8 MB
  h16*   wuph    = (h16*)(ws + 40 * MB);      // 8 MB
  h16*   wdownh  = (h16*)(ws + 48 * MB);      // 8 MB
  char*  big     = ws + 56 * MB;
  float* qkv     = (float*)big;               // 48 MB (phase A)
  h16*   qh      = (h16*)(big + 48 * MB);     // 8 MB
  h16*   kh      = (h16*)(big + 56 * MB);     // 8 MB
  h16*   vth     = (h16*)(big + 64 * MB);     // 8 MB
  h16*   oh      = (h16*)(big + 72 * MB);     // 8 MB
  float* gatef   = (float*)big;               // 64 MB (phase B, reuses qkv..kh)
  h16*   gatedh  = (h16*)(big + 64 * MB);     // 32 MB (reuses vth/oh)

  auto cvt = [&](const float* src, h16* dst, int n) {
    cvt_f32_f16<<<dim3((n / 4 + 255) / 256), dim3(256), 0, stream>>>(src, dst, n);
  };
  cvt(w_qkv,  wqkvh,  D * 3 * D);
  cvt(w_out,  wouth,  D * D);
  cvt(w_gate, wgateh, D * F);
  cvt(w_up,   wuph,   D * F);
  cvt(w_down, wdownh, F * D);

  // ---- attention sub-block ----
  layernorm_f16<<<dim3(R), dim3(256), 0, stream>>>(x, g1, b1, ybf);
  gemm_f16<EPI_PLAIN><<<dim3(3 * D / 128, R / 128), dim3(256), 0, stream>>>(
      ybf, wqkvh, 3 * D, D, nullptr, nullptr, nullptr, qkv, nullptr);
  rope_split<<<dim3(R * 16 * 32 / 256), dim3(256), 0, stream>>>(qkv, qh, kh, vth);
  attn_local<<<dim3(32, S / 64), dim3(128), 0, stream>>>(qh, kh, vth, oh);
  gemm_f16<EPI_RESID><<<dim3(D / 128, R / 128), dim3(256), 0, stream>>>(
      oh, wouth, D, D, nullptr, nullptr, x, x1, nullptr);

  // ---- SwiGLU MLP sub-block ----
  layernorm_f16<<<dim3(R), dim3(256), 0, stream>>>(x1, g2, b2, ybf);
  gemm_f16<EPI_BIAS><<<dim3(F / 128, R / 128), dim3(256), 0, stream>>>(
      ybf, wgateh, F, D, b_gate, nullptr, nullptr, gatef, nullptr);
  gemm_f16<EPI_SWIGLU><<<dim3(F / 128, R / 128), dim3(256), 0, stream>>>(
      ybf, wuph, F, D, b_up, gatef, nullptr, nullptr, gatedh);
  gemm_f16<EPI_DOWN><<<dim3(D / 128, R / 128), dim3(256), 0, stream>>>(
      gatedh, wdownh, D, F, b_down, nullptr, x1, out, nullptr);
}
